// RecurrentActorCritic_11269994184960
// MI455X (gfx1250) — matrix operand order repaired and checked by reading, not compiled, not run on hardware
//
#include <hip/hip_runtime.h>
#include <hip/hip_bf16.h>

typedef __attribute__((ext_vector_type(16))) _Float16 v16h;
typedef __attribute__((ext_vector_type(8)))  _Float16 v8h;
typedef __attribute__((ext_vector_type(8)))  float    v8f;

constexpr int Bsz = 32768;   // batch
constexpr int SD  = 512;     // STATE_DIM
constexpr int HID = 256;     // HIDDEN
constexpr int AH  = 128;     // ACT_HID
constexpr int AD  = 128;     // ACTION_DIM

// ---- WMMA fragment loaders -------------------------------------------------
// A fragment 16x32 (MxK), f32 source converted on the fly:
// lane L: row = L&15; elems 0..7 -> K = k + (L>>4)*8 .., elems 8..15 -> +16.
__device__ __forceinline__ v16h load_af(const float* __restrict__ base, int ld,
                                        int row, int k, int lane) {
  const float* p = base + (size_t)(row + (lane & 15)) * ld + k + ((lane >> 4) << 3);
  v16h r;
#pragma unroll
  for (int e = 0; e < 8; ++e) r[e] = (_Float16)p[e];
#pragma unroll
  for (int e = 0; e < 8; ++e) r[8 + e] = (_Float16)p[16 + e];
  return r;
}

// B fragment 32x16 (KxN) for x @ W^T: column n of B = row (col0+n) of W.
// lane L: elems 0..7 -> K = k + (L>>4)*8 .., elems 8..15 -> +16 (same as A).
__device__ __forceinline__ v16h load_bf(const float* __restrict__ base, int ld,
                                        int col0, int k, int lane) {
  const float* p = base + (size_t)(col0 + (lane & 15)) * ld + k + ((lane >> 4) << 3);
  v16h r;
#pragma unroll
  for (int e = 0; e < 8; ++e) r[e] = (_Float16)p[e];
#pragma unroll
  for (int e = 0; e < 8; ++e) r[8 + e] = (_Float16)p[16 + e];
  return r;
}

// f16-weight version: two runs of 8 halves (K +0..7, +16..23), 16B-aligned -> 2x b128 loads.
__device__ __forceinline__ v16h load_bf(const _Float16* __restrict__ base, int ld,
                                        int col0, int k, int lane) {
  const _Float16* p = base + (size_t)(col0 + (lane & 15)) * ld + k + ((lane >> 4) << 3);
  const v8h lo = *(const v8h*)p;
  const v8h hi = *(const v8h*)(p + 16);
  v16h r;
#pragma unroll
  for (int e = 0; e < 8; ++e) { r[e] = lo[e]; r[8 + e] = hi[e]; }
  return r;
}

__device__ __forceinline__ v8f wmma_f16(v16h a, v16h b, v8f c) {
  return __builtin_amdgcn_wmma_f32_16x16x32_f16(
      /*neg_a=*/false, a, /*neg_b=*/false, b,
      /*c_mod=*/(short)0, c, /*reuse_a=*/false, /*reuse_b=*/false);
}

__device__ __forceinline__ float sigm(float x) { return 1.0f / (1.0f + expf(-x)); }

// ---- one-shot f32 -> f16 weight conversion (n multiple of 8) ---------------
__global__ __launch_bounds__(256) void cvt_f16_kernel(
    const float* __restrict__ src, _Float16* __restrict__ dst, int n) {
  const int i = (blockIdx.x * 256 + threadIdx.x) * 8;
  if (i >= n) return;
  v8h o;
#pragma unroll
  for (int e = 0; e < 8; ++e) o[e] = (_Float16)src[i + e];
  *(v8h*)(dst + i) = o;
}

// ============================================================================
// Kernel A: fused LSTM cell.  One wave: 32 batch rows x 16 hidden cols,
// computing i/f/g/o tiles together so the cell update stays in registers.
// ============================================================================
template <typename WT>
__global__ __launch_bounds__(128) void lstm_cell_kernel(
    const float* __restrict__ state, const float* __restrict__ h0,
    const float* __restrict__ c0,
    const WT* __restrict__ W_ih, const WT* __restrict__ W_hh,
    const float* __restrict__ b_ih, const float* __restrict__ b_hh,
    float* __restrict__ h1, float* __restrict__ c1) {
  const int lane = threadIdx.x & 31;
  const int task = blockIdx.x * 4 + (threadIdx.x >> 5); // 4 waves/block share rows
  const int r0 = (task >> 4) * 32;   // batch row block (Bsz/32 = 1024)
  const int j0 = (task & 15) * 16;   // hidden col tile  (HID/16 = 16)

  v8f acc[4][2];
#pragma unroll
  for (int g = 0; g < 4; ++g) {
    const int j = g * HID + j0 + (lane & 15);
    const float bias = b_ih[j] + b_hh[j];
#pragma unroll
    for (int t = 0; t < 2; ++t)
#pragma unroll
      for (int v = 0; v < 8; ++v) acc[g][t][v] = bias;
  }

  // gates += state @ W_ih^T
  for (int k = 0; k < SD; k += 32) {
    const v16h a0 = load_af(state, SD, r0, k, lane);
    const v16h a1 = load_af(state, SD, r0 + 16, k, lane);
#pragma unroll
    for (int g = 0; g < 4; ++g) {
      const v16h b = load_bf(W_ih, SD, g * HID + j0, k, lane);
      acc[g][0] = wmma_f16(a0, b, acc[g][0]);
      acc[g][1] = wmma_f16(a1, b, acc[g][1]);
    }
  }
  // gates += h0 @ W_hh^T
  for (int k = 0; k < HID; k += 32) {
    const v16h a0 = load_af(h0, HID, r0, k, lane);
    const v16h a1 = load_af(h0, HID, r0 + 16, k, lane);
#pragma unroll
    for (int g = 0; g < 4; ++g) {
      const v16h b = load_bf(W_hh, HID, g * HID + j0, k, lane);
      acc[g][0] = wmma_f16(a0, b, acc[g][0]);
      acc[g][1] = wmma_f16(a1, b, acc[g][1]);
    }
  }

  // Cell update straight from accumulators. C/D layout: lane L, VGPR v ->
  // M = v + 8*(L>=16), N = L&15.
  const int n = lane & 15;
  const int mhi = (lane >> 4) * 8;
#pragma unroll
  for (int t = 0; t < 2; ++t) {
#pragma unroll
    for (int v = 0; v < 8; ++v) {
      const int row = r0 + t * 16 + mhi + v;
      const size_t idx = (size_t)row * HID + j0 + n;
      const float iv = sigm(acc[0][t][v]);
      const float fv = sigm(acc[1][t][v]);
      const float gv = tanhf(acc[2][t][v]);
      const float ov = sigm(acc[3][t][v]);
      const float cn = fv * c0[idx] + iv * gv;
      c1[idx] = cn;
      h1[idx] = ov * tanhf(cn);
    }
  }
}

// ============================================================================
// Kernel B: fused actor head + Gumbel-max sampling.
// Block = 16 batch rows, 8 waves = 8 column tiles of 16.
// ============================================================================
__device__ __forceinline__ float gumbel_hash(uint32_t row, uint32_t col) {
  uint32_t x = row * 0x9E3779B9u + col * 0x85EBCA6Bu + 42u;
  x ^= x >> 16; x *= 0x7FEB352Du; x ^= x >> 15; x *= 0x846CA68Bu; x ^= x >> 16;
  const float u = (float)(x >> 8) * (1.0f / 16777216.0f) + 1.0e-12f;
  return -logf(-logf(u));
}

template <typename WT>
__global__ __launch_bounds__(256) void actor_head_kernel(
    const float* __restrict__ h1, const WT* __restrict__ Wa1,
    const float* __restrict__ ba1, const WT* __restrict__ Wa2,
    const float* __restrict__ ba2, const float* __restrict__ mask,
    float* __restrict__ logits_out, float* __restrict__ action_out) {
  __shared__ _Float16 a_lds[16][AH + 8]; // relu(h1@Wa1^T+b) in f16, A-layout
  __shared__ float    lg_lds[16][AD];    // logits + gumbel for argmax

  const int lane = threadIdx.x & 31;
  const int wv = threadIdx.x >> 5; // 0..7
  const int r0 = blockIdx.x * 16;
  const int n = lane & 15;
  const int mhi = (lane >> 4) * 8;

  // ---- stage 1: a = relu(h1 @ Wa1^T + ba1), this wave's 16 cols
  {
    const int c0a = wv * 16;
    v8f acc;
    const float bias = ba1[c0a + n];
#pragma unroll
    for (int v = 0; v < 8; ++v) acc[v] = bias;
    for (int k = 0; k < HID; k += 32) {
      const v16h a = load_af(h1, HID, r0, k, lane);
      const v16h b = load_bf(Wa1, HID, c0a, k, lane);
      acc = wmma_f16(a, b, acc);
    }
#pragma unroll
    for (int v = 0; v < 8; ++v)
      a_lds[mhi + v][c0a + n] = (_Float16)fmaxf(acc[v], 0.0f);
  }
  __syncthreads();

  // ---- stage 2: logits = a @ Wa2^T + ba2 + log(mask + eps)
  {
    const int c0b = wv * 16;
    v8f acc;
    const float bias = ba2[c0b + n];
#pragma unroll
    for (int v = 0; v < 8; ++v) acc[v] = bias;
#pragma unroll
    for (int k = 0; k < AH; k += 32) {
      const int m = lane & 15;
      const int kb = k + ((lane >> 4) << 3);
      // 16B-aligned vector loads from LDS (row stride 272B, kb multiple of 8)
      const v8h lo = *(const v8h*)&a_lds[m][kb];
      const v8h hi = *(const v8h*)&a_lds[m][kb + 16];
      v16h a;
#pragma unroll
      for (int e = 0; e < 8; ++e) { a[e] = lo[e]; a[8 + e] = hi[e]; }
      const v16h b = load_bf(Wa2, AH, c0b, k, lane);
      acc = wmma_f16(a, b, acc);
    }
#pragma unroll
    for (int v = 0; v < 8; ++v) {
      const int m = mhi + v;
      const int row = r0 + m;
      const int col = c0b + n;
      const float lg = acc[v] + logf(mask[(size_t)row * AD + col] + 1e-8f);
      logits_out[(size_t)row * AD + col] = lg;
      lg_lds[m][col] = lg + gumbel_hash((uint32_t)row, (uint32_t)col);
    }
  }
  __syncthreads();

  // ---- Gumbel-max argmax, one thread per row
  if (threadIdx.x < 16) {
    const int m = threadIdx.x;
    float best = lg_lds[m][0];
    int bi = 0;
#pragma unroll 4
    for (int c = 1; c < AD; ++c) {
      const float v = lg_lds[m][c];
      if (v > best) { best = v; bi = c; }
    }
    action_out[r0 + m] = (float)bi;
  }
}

// ============================================================================
extern "C" void kernel_launch(void* const* d_in, const int* in_sizes, int n_in,
                              void* d_out, int out_size, void* d_ws, size_t ws_size,
                              hipStream_t stream) {
  (void)in_sizes; (void)n_in; (void)out_size;
  const float* state = (const float*)d_in[0];
  const float* mask  = (const float*)d_in[1];
  const float* h0    = (const float*)d_in[2];
  const float* c0    = (const float*)d_in[3];
  const float* W_ih  = (const float*)d_in[4];
  const float* W_hh  = (const float*)d_in[5];
  const float* b_ih  = (const float*)d_in[6];
  const float* b_hh  = (const float*)d_in[7];
  const float* Wa1   = (const float*)d_in[8];
  const float* ba1   = (const float*)d_in[9];
  const float* Wa2   = (const float*)d_in[10];
  const float* ba2   = (const float*)d_in[11];

  float* out = (float*)d_out; // return order: action, logits, h1, c1
  float* d_action = out;
  float* d_logits = out + (size_t)Bsz;
  float* d_h1 = d_logits + (size_t)Bsz * AD;
  float* d_c1 = d_h1 + (size_t)Bsz * HID;

  const int blocksA = (Bsz / 32) * (HID / 16) / 4; // 16384 wave-tasks, 4/block
  const int blocksB = Bsz / 16;

  constexpr size_t N_WIH = (size_t)4 * HID * SD;   // 524288
  constexpr size_t N_WHH = (size_t)4 * HID * HID;  // 262144
  constexpr size_t N_WA1 = (size_t)AH * HID;       // 32768
  constexpr size_t N_WA2 = (size_t)AD * AH;        // 16384
  constexpr size_t N_W   = N_WIH + N_WHH + N_WA1 + N_WA2;

  if (ws_size >= N_W * sizeof(_Float16)) {
    // Pre-convert weights to f16 once: halves weight L2 traffic and removes
    // per-k-step v_cvt from the WMMA hot loop.
    _Float16* wih16 = (_Float16*)d_ws;
    _Float16* whh16 = wih16 + N_WIH;
    _Float16* wa116 = whh16 + N_WHH;
    _Float16* wa216 = wa116 + N_WA1;
    cvt_f16_kernel<<<dim3(N_WIH / 2048), dim3(256), 0, stream>>>(W_ih, wih16, (int)N_WIH);
    cvt_f16_kernel<<<dim3(N_WHH / 2048), dim3(256), 0, stream>>>(W_hh, whh16, (int)N_WHH);
    cvt_f16_kernel<<<dim3(N_WA1 / 2048), dim3(256), 0, stream>>>(Wa1, wa116, (int)N_WA1);
    cvt_f16_kernel<<<dim3(N_WA2 / 2048), dim3(256), 0, stream>>>(Wa2, wa216, (int)N_WA2);
    lstm_cell_kernel<_Float16><<<dim3(blocksA), dim3(128), 0, stream>>>(
        state, h0, c0, wih16, whh16, b_ih, b_hh, d_h1, d_c1);
    actor_head_kernel<_Float16><<<dim3(blocksB), dim3(256), 0, stream>>>(
        d_h1, wa116, ba1, wa216, ba2, mask, d_logits, d_action);
  } else {
    // Fallback: convert weights in-loop (still WMMA).
    lstm_cell_kernel<float><<<dim3(blocksA), dim3(128), 0, stream>>>(
        state, h0, c0, W_ih, W_hh, b_ih, b_hh, d_h1, d_c1);
    actor_head_kernel<float><<<dim3(blocksB), dim3(256), 0, stream>>>(
        d_h1, Wa1, ba1, Wa2, ba2, mask, d_logits, d_action);
  }
}
